// PPOAgent_24283745092125
// MI455X (gfx1250) — matrix operand order repaired and checked by reading, not compiled, not run on hardware
//
#include <hip/hip_runtime.h>
#include <math.h>

// ---------------------------------------------------------------------------
// PPO loss for MI455X (gfx1250, wave32, WMMA, TDM).
// GEMMs run on v_wmma_f32_16x16x32_bf16; activations kept in bf16; bf16-input
// GEMMs stage their A-tiles with TENSOR_LOAD_TO_LDS (TDM) when available.
// ---------------------------------------------------------------------------

#define T_STEPS 128
#define B_BATCH 2048
#define D_OBS   256
#define A_ACT   32
#define H_DIM   512
#define NP      (T_STEPS * B_BATCH)          // 262144 policy rows
#define NV      ((T_STEPS + 1) * B_BATCH)    // 264192 value rows

typedef __attribute__((ext_vector_type(16))) __bf16 v16bf;
typedef __attribute__((ext_vector_type(8)))  float  v8f;
typedef unsigned int u32x4 __attribute__((ext_vector_type(4)));
typedef int          i32x4 __attribute__((ext_vector_type(4)));
typedef int          i32x8 __attribute__((ext_vector_type(8)));

union Frag16 { unsigned int u[8]; v16bf v; };

#if defined(__gfx1250__) && __has_builtin(__builtin_amdgcn_tensor_load_to_lds) && \
    __has_builtin(__builtin_amdgcn_s_wait_tensorcnt)
#define USE_TDM 1
#else
#define USE_TDM 0
#endif

// native f32 -> bf16 (RNE) pack; lowers to v_cvt_pk_bf16_f32 on gfx1250
__device__ __forceinline__ unsigned int pk2f(float a, float b) {
  union { __bf16 h[2]; unsigned int u; } c;
  c.h[0] = (__bf16)a; c.h[1] = (__bf16)b;
  return c.u;
}

#if USE_TDM
// Issue a TDM load of a [128 rows x 32 cols] bf16 tile (row stride K elems)
// into LDS at byte offset ldsOff. D# layout per CDNA5 ISA.
__device__ __forceinline__ void tdm_load_tile_bf16(unsigned int ldsOff,
                                                   const __bf16* g,
                                                   unsigned int K) {
  unsigned long long ga = (unsigned long long)g;
  u32x4 g0;
  g0[0] = 1u;                                    // count=1, user descriptor
  g0[1] = ldsOff;                                // lds_addr (bytes)
  g0[2] = (unsigned int)ga;                      // global_addr[31:0]
  g0[3] = ((unsigned int)(ga >> 32) & 0x01FFFFFFu) | (2u << 30);  // addr hi | type=2
  i32x8 g1;
  g1[0] = 0x00010000;                            // data_size=1 (2B), no multicast
  g1[1] = (int)((K & 0xFFFFu) << 16);            // tensor_dim0[15:0]
  g1[2] = (int)((K >> 16) & 0xFFFFu);            // tensor_dim0[31:16] | tdim1 lo=0
  g1[3] = 0x10 | (32 << 16);                     // tensor_dim1 hi (1M rows) | tile_dim0=32
  g1[4] = 128;                                   // tile_dim1=128, tile_dim2=0
  g1[5] = (int)K;                                // tensor_dim0_stride[31:0]
  g1[6] = 0;                                     // stride hi, dim1_stride lo
  g1[7] = 0;
  i32x4 z4 = {0, 0, 0, 0};
#if __clang_major__ >= 23
  i32x8 z8 = {0, 0, 0, 0, 0, 0, 0, 0};
  __builtin_amdgcn_tensor_load_to_lds(g0, g1, z4, z4, z8, 0);
#else
  __builtin_amdgcn_tensor_load_to_lds(g0, g1, z4, z4, 0);
#endif
}
#endif

// ---------------------------------------------------------------------------
// Layer-1 GEMM (f32 input):  Y[N x M] = silu( X[N x K] @ W[K x M] + bias )
// 256 threads = 8 wave32s; block tile 128x64; wave tile 32x32 (2x2 frags).
// LDS pre-packed in the CDNA5 WMMA A(16x32 bf16)/B(32x16 bf16) lane layout.
// ---------------------------------------------------------------------------
__global__ __launch_bounds__(256)
void gemm_l1(const float* __restrict__ X, const float* __restrict__ W,
             const float* __restrict__ bias, __bf16* __restrict__ Y,
             int K, int M) {
  const int t    = threadIdx.x;
  const int lane = t & 31;
  const int wave = t >> 5;
  const long rowBase = (long)blockIdx.x * 128;
  const int  colBase = blockIdx.y * 64;

  __shared__ unsigned int ldsA[8 * 32 * 8];   // [mchunk][lane][v]  8 KB
  __shared__ unsigned int ldsB[4 * 32 * 8];   // [nchunk][lane][v]  4 KB

  v8f acc[2][2] = {};

  const int a_mchunk = t >> 5;
  const int a_lane   = t & 31;
  const int a_m      = a_mchunk * 16 + (a_lane & 15);
  const int a_kgrp   = a_lane >> 4;
  const float* arow  = X + (rowBase + a_m) * (long)K;

  const int wm = wave >> 1;
  const int wn = wave & 1;

  for (int k0 = 0; k0 < K; k0 += 32) {
#pragma unroll
    for (int v = 0; v < 8; ++v) {
      int k = (v < 4) ? (a_kgrp * 8 + 2 * v) : (16 + a_kgrp * 8 + 2 * (v - 4));
      const float* p = arow + k0 + k;
      ldsA[(a_mchunk * 32 + a_lane) * 8 + v] = pk2f(p[0], p[1]);
    }
#pragma unroll
    for (int j = 0; j < 4; ++j) {
      int flat   = t * 4 + j;
      int nchunk = flat >> 8;
      int bl     = (flat >> 3) & 31;
      int v      = flat & 7;
      int kb     = (v < 4) ? ((bl >> 4) * 8 + 2 * v) : (16 + (bl >> 4) * 8 + 2 * (v - 4));
      int n      = nchunk * 16 + (bl & 15);
      const float* wp = W + (long)(k0 + kb) * M + colBase + n;
      ldsB[flat] = pk2f(wp[0], wp[M]);
    }
    __syncthreads();
    if (k0 + 32 < K) __builtin_prefetch(arow + k0 + 32, 0, 0);  // global_prefetch_b8

    Frag16 a[2], b[2];
#pragma unroll
    for (int i = 0; i < 2; ++i) {
      const unsigned int* src = &ldsA[((wm * 2 + i) * 32 + lane) * 8];
#pragma unroll
      for (int v = 0; v < 8; ++v) a[i].u[v] = src[v];
    }
#pragma unroll
    for (int j = 0; j < 2; ++j) {
      const unsigned int* src = &ldsB[((wn * 2 + j) * 32 + lane) * 8];
#pragma unroll
      for (int v = 0; v < 8; ++v) b[j].u[v] = src[v];
    }
#pragma unroll
    for (int i = 0; i < 2; ++i)
#pragma unroll
      for (int j = 0; j < 2; ++j)
        acc[i][j] = __builtin_amdgcn_wmma_f32_16x16x32_bf16(
            false, a[i].v, false, b[j].v, (short)0, acc[i][j], false, false);
    __syncthreads();
  }

  const int cm = lane >> 4;
  const int cn = lane & 15;
#pragma unroll
  for (int i = 0; i < 2; ++i)
#pragma unroll
    for (int j = 0; j < 2; ++j) {
      int ng = colBase + wn * 32 + j * 16 + cn;
      float bv = bias[ng];
#pragma unroll
      for (int r = 0; r < 8; ++r) {
        long mg = rowBase + wm * 32 + i * 16 + cm * 8 + r;
        float val = acc[i][j][r] + bv;
        val = val / (1.0f + __expf(-val));            // SiLU
        Y[mg * M + ng] = (__bf16)val;
      }
    }
}

// ---------------------------------------------------------------------------
// bf16-activation GEMM (layers 2/3). A-tile staged raw (row-major 128x32 bf16,
// 8KB) by the Tensor Data Mover when available; fragments still read as two
// contiguous ds_load_b128 per operand (ISA A-layout: v0..3 / v4..7 contiguous).
// ---------------------------------------------------------------------------
template <bool DO_SILU, bool OUT_BF16>
__global__ __launch_bounds__(256)
void gemm_act(const __bf16* __restrict__ X, const float* __restrict__ W,
              const float* __restrict__ bias, void* __restrict__ Yv,
              int K, int M) {
  const int t    = threadIdx.x;
  const int lane = t & 31;
  const int wave = t >> 5;
  const long rowBase = (long)blockIdx.x * 128;
  const int  colBase = blockIdx.y * 64;

  __shared__ __align__(16) unsigned int ldsRawA[128 * 16];  // 128 rows x 32 bf16
  __shared__ unsigned int ldsB[4 * 32 * 8];

  v8f acc[2][2] = {};
  const int wm = wave >> 1;
  const int wn = wave & 1;

#if USE_TDM
  // Low 32 bits of a generic LDS pointer are the LDS byte offset (ISA 10.2).
  const unsigned int ldsOff = (unsigned int)(unsigned long long)(const void*)ldsRawA;
#endif

  for (int k0 = 0; k0 < K; k0 += 32) {
#if USE_TDM
    if (wave == 0)
      tdm_load_tile_bf16(ldsOff, X + rowBase * (long)K + k0, (unsigned int)K);
#else
    {
      int row = t >> 1;
      const unsigned int* src =
          (const unsigned int*)(X + (rowBase + row) * (long)K + k0) + (t & 1) * 8;
      unsigned int* dst = &ldsRawA[row * 16 + (t & 1) * 8];
#pragma unroll
      for (int j = 0; j < 8; ++j) dst[j] = src[j];
    }
#endif
#pragma unroll
    for (int j = 0; j < 4; ++j) {
      int flat   = t * 4 + j;
      int nchunk = flat >> 8;
      int bl     = (flat >> 3) & 31;
      int v      = flat & 7;
      int kb     = (v < 4) ? ((bl >> 4) * 8 + 2 * v) : (16 + (bl >> 4) * 8 + 2 * (v - 4));
      int n      = nchunk * 16 + (bl & 15);
      const float* wp = W + (long)(k0 + kb) * M + colBase + n;
      ldsB[flat] = pk2f(wp[0], wp[M]);
    }
#if USE_TDM
    __builtin_amdgcn_s_wait_tensorcnt(0);
#endif
    __syncthreads();
    if (k0 + 32 < K) __builtin_prefetch(X + rowBase * (long)K + k0 + 32, 0, 0);

    Frag16 a[2], b[2];
#pragma unroll
    for (int i = 0; i < 2; ++i) {
      int m = (wm * 2 + i) * 16 + (lane & 15);
      const unsigned int* base = &ldsRawA[m * 16 + 4 * (lane >> 4)];
#pragma unroll
      for (int v = 0; v < 4; ++v) a[i].u[v] = base[v];        // K=0..7 half
#pragma unroll
      for (int v = 0; v < 4; ++v) a[i].u[4 + v] = base[8 + v]; // K=16..23 half
    }
#pragma unroll
    for (int j = 0; j < 2; ++j) {
      const unsigned int* src = &ldsB[((wn * 2 + j) * 32 + lane) * 8];
#pragma unroll
      for (int v = 0; v < 8; ++v) b[j].u[v] = src[v];
    }
#pragma unroll
    for (int i = 0; i < 2; ++i)
#pragma unroll
      for (int j = 0; j < 2; ++j)
        acc[i][j] = __builtin_amdgcn_wmma_f32_16x16x32_bf16(
            false, a[i].v, false, b[j].v, (short)0, acc[i][j], false, false);
    __syncthreads();
  }

  const int cm = lane >> 4;
  const int cn = lane & 15;
#pragma unroll
  for (int i = 0; i < 2; ++i)
#pragma unroll
    for (int j = 0; j < 2; ++j) {
      int ng = colBase + wn * 32 + j * 16 + cn;
      float bv = bias[ng];
#pragma unroll
      for (int r = 0; r < 8; ++r) {
        long mg = rowBase + wm * 32 + i * 16 + cm * 8 + r;
        float val = acc[i][j][r] + bv;
        if (DO_SILU) val = val / (1.0f + __expf(-val));
        if (OUT_BF16) ((__bf16*)Yv)[mg * M + ng] = (__bf16)val;
        else          ((float*)Yv)[mg * M + ng] = val;
      }
    }
}

// ---------------------------------------------------------------------------
// Value head GEMV: baseline[r] = h2[r,:] . vW3 + vb3  (K=512, M=1)
// ---------------------------------------------------------------------------
__global__ __launch_bounds__(256)
void gemv_head(const __bf16* __restrict__ H, const float* __restrict__ w,
               const float* __restrict__ b, float* __restrict__ out, int rows) {
  int wave = threadIdx.x >> 5, lane = threadIdx.x & 31;
  long row = (long)blockIdx.x * 8 + wave;
  if (row >= rows) return;
  const __bf16* hp = H + row * H_DIM;
  float s = 0.f;
#pragma unroll 4
  for (int i = lane; i < H_DIM; i += 32) s += (float)hp[i] * w[i];
#pragma unroll
  for (int off = 16; off > 0; off >>= 1) s += __shfl_down(s, off, 32);
  if (lane == 0) out[row] = s + b[0];
}

// ---------------------------------------------------------------------------
// GAE reverse scan (thread per batch column).
// ---------------------------------------------------------------------------
__global__ __launch_bounds__(256)
void gae_kernel(const float* __restrict__ reward, const int* __restrict__ done,
                const int* __restrict__ trunc, const float* __restrict__ baseline,
                float* __restrict__ vs, float* __restrict__ adv) {
  const float G = 0.99f, LAM = 0.95f;
  int b = blockIdx.x * blockDim.x + threadIdx.x;
  if (b >= B_BATCH) return;
  float bootstrap = baseline[(long)T_STEPS * B_BATCH + b];
  float acc = 0.f;
  for (int t = T_STEPS - 1; t >= 0; --t) {
    long idx = (long)t * B_BATCH + b;
    float tr = (float)trunc[idx], dn = (float)done[idx];
    float tmask = 1.f - tr;
    float term  = dn * (1.f - tr);
    float v     = baseline[idx];
    float vtp1  = (t == T_STEPS - 1) ? bootstrap : baseline[idx + B_BATCH];
    float delta = (reward[idx] + G * (1.f - term) * vtp1 - v) * tmask;
    float fac   = G * (1.f - term) * tmask * LAM;
    acc = delta + fac * acc;
    vs[idx] = acc + v;
  }
  for (int t = 0; t < T_STEPS; ++t) {
    long idx = (long)t * B_BATCH + b;
    float tr = (float)trunc[idx], dn = (float)done[idx];
    float tmask = 1.f - tr;
    float term  = dn * (1.f - tr);
    float vstp1 = (t == T_STEPS - 1) ? bootstrap : vs[idx + B_BATCH];
    adv[idx] = (reward[idx] + G * (1.f - term) * vstp1 - baseline[idx]) * tmask;
  }
}

// ---------------------------------------------------------------------------
// Per-element loss + deterministic two-stage reduction.
// ---------------------------------------------------------------------------
__device__ __forceinline__ float softplusf(float x) {
  return (x > 15.f) ? x : log1pf(__expf(x));
}
__device__ __forceinline__ float ldjf(float x) {
  return 2.f * (0.6931471805599453f - x - softplusf(-2.f * x));
}
__device__ __forceinline__ float hash_normal(unsigned int idx) {
  unsigned int s = idx * 747796405u + 2891336453u;
  s = ((s >> ((s >> 28) + 4u)) ^ s) * 277803737u;
  s = (s >> 22) ^ s;
  unsigned int s2 = (idx ^ 0x9E3779B9u) * 2654435761u + 1013904223u;
  s2 ^= s2 << 13; s2 ^= s2 >> 17; s2 ^= s2 << 5;
  float u1 = ((s  & 0xFFFFFFu) + 1u) * (1.0f / 16777217.0f);
  float u2 = (s2 & 0xFFFFFFu) * (1.0f / 16777216.0f);
  return __fsqrt_rn(-2.f * __logf(u1)) * __cosf(6.2831853071795865f * u2);
}

__global__ __launch_bounds__(256)
void loss_kernel(const float* __restrict__ logits, const float* __restrict__ blogits,
                 const float* __restrict__ action, const float* __restrict__ vs,
                 const float* __restrict__ adv, const float* __restrict__ baseline,
                 float* __restrict__ partials) {
  const float HLP = 0.9189385332046727f;
  long idx = (long)blockIdx.x * 256 + threadIdx.x;

  float tgt_lp = 0.f, beh_lp = 0.f, ent = 0.f;
  const float* lg = logits  + idx * (2 * A_ACT);
  const float* bg = blogits + idx * (2 * A_ACT);
  const float* ac = action  + idx * A_ACT;
#pragma unroll 4
  for (int a = 0; a < A_ACT; ++a) {
    float act   = ac[a];
    float loc   = lg[a];
    float scale = softplusf(lg[A_ACT + a]) + 0.001f;
    float z     = (act - loc) / scale;
    float lsc   = __logf(scale);
    tgt_lp += -0.5f * z * z - (HLP + lsc) - ldjf(act);

    float bloc   = bg[a];
    float bscale = softplusf(bg[A_ACT + a]) + 0.001f;
    float bz     = (act - bloc) / bscale;
    beh_lp += -0.5f * bz * bz - (HLP + __logf(bscale)) - ldjf(act);

    float smp = loc + scale * hash_normal((unsigned int)(idx * A_ACT + a));
    ent += 0.5f + HLP + lsc + ldjf(smp);
  }
  float rho = __expf(tgt_lp - beh_lp);
  float a_  = adv[idx];
  float s1  = rho * a_;
  float s2  = fminf(fmaxf(rho, 0.7f), 1.3f) * a_;
  float verr = vs[idx] - baseline[idx];
  float contrib = -fminf(s1, s2) + 0.25f * verr * verr - 0.01f * ent;

  __shared__ float sdata[256];
  sdata[threadIdx.x] = contrib;
  __syncthreads();
#pragma unroll
  for (int s = 128; s > 0; s >>= 1) {
    if (threadIdx.x < s) sdata[threadIdx.x] += sdata[threadIdx.x + s];
    __syncthreads();
  }
  if (threadIdx.x == 0) partials[blockIdx.x] = sdata[0];
}

__global__ __launch_bounds__(256)
void finalize_kernel(const float* __restrict__ partials, int n, float* __restrict__ out) {
  __shared__ float sdata[256];
  float s = 0.f;
  for (int i = threadIdx.x; i < n; i += 256) s += partials[i];
  sdata[threadIdx.x] = s;
  __syncthreads();
#pragma unroll
  for (int k = 128; k > 0; k >>= 1) {
    if (threadIdx.x < k) sdata[threadIdx.x] += sdata[threadIdx.x + k];
    __syncthreads();
  }
  if (threadIdx.x == 0) out[0] = sdata[0] * (1.0f / (float)NP);
}

// ---------------------------------------------------------------------------
// Launch
// ---------------------------------------------------------------------------
extern "C" void kernel_launch(void* const* d_in, const int* in_sizes, int n_in,
                              void* d_out, int out_size, void* d_ws, size_t ws_size,
                              hipStream_t stream) {
  const float* obs   = (const float*)d_in[0];
  const float* rew   = (const float*)d_in[1];
  const int*   done  = (const int*)  d_in[2];
  const int*   trunc = (const int*)  d_in[3];
  const float* act   = (const float*)d_in[4];
  const float* blog  = (const float*)d_in[5];
  const float* pW1 = (const float*)d_in[6],  *pb1 = (const float*)d_in[7];
  const float* pW2 = (const float*)d_in[8],  *pb2 = (const float*)d_in[9];
  const float* pW3 = (const float*)d_in[10], *pb3 = (const float*)d_in[11];
  const float* vW1 = (const float*)d_in[12], *vb1 = (const float*)d_in[13];
  const float* vW2 = (const float*)d_in[14], *vb2 = (const float*)d_in[15];
  const float* vW3 = (const float*)d_in[16], *vb3 = (const float*)d_in[17];

  char* ws = (char*)d_ws;
  const size_t SZ_H   = (size_t)NV * H_DIM * sizeof(unsigned short);
  const size_t SZ_LOG = (size_t)NP * 2 * A_ACT * sizeof(float);
  const size_t SZ_BAS = (size_t)NV * sizeof(float);
  const size_t SZ_TB  = (size_t)NP * sizeof(float);
  __bf16* hA      = (__bf16*)(ws);
  __bf16* hB      = (__bf16*)(ws + SZ_H);
  float* logits   = (float*)(ws + 2 * SZ_H);
  float* baseline = (float*)(ws + 2 * SZ_H + SZ_LOG);
  float* vsb      = (float*)(ws + 2 * SZ_H + SZ_LOG + SZ_BAS);
  float* advb     = (float*)(ws + 2 * SZ_H + SZ_LOG + SZ_BAS + SZ_TB);
  float* partials = (float*)(ws + 2 * SZ_H + SZ_LOG + SZ_BAS + 2 * SZ_TB);

  const dim3 blk(256);

  // policy MLP
  gemm_l1<<<dim3(NP / 128, H_DIM / 64), blk, 0, stream>>>(obs, pW1, pb1, hA, D_OBS, H_DIM);
  gemm_act<true, true><<<dim3(NP / 128, H_DIM / 64), blk, 0, stream>>>(hA, pW2, pb2, hB, H_DIM, H_DIM);
  gemm_act<false, false><<<dim3(NP / 128, 1), blk, 0, stream>>>(hB, pW3, pb3, logits, H_DIM, 2 * A_ACT);

  // value MLP
  gemm_l1<<<dim3(NV / 128, H_DIM / 64), blk, 0, stream>>>(obs, vW1, vb1, hA, D_OBS, H_DIM);
  gemm_act<true, true><<<dim3(NV / 128, H_DIM / 64), blk, 0, stream>>>(hA, vW2, vb2, hB, H_DIM, H_DIM);
  gemv_head<<<dim3(NV / 8), blk, 0, stream>>>(hB, vW3, vb3, baseline, NV);

  // GAE + losses
  gae_kernel<<<dim3(B_BATCH / 256), blk, 0, stream>>>(rew, done, trunc, baseline, vsb, advb);
  loss_kernel<<<dim3(NP / 256), blk, 0, stream>>>(logits, blog, act, vsb, advb, baseline, partials);
  finalize_kernel<<<dim3(1), blk, 0, stream>>>(partials, NP / 256, (float*)d_out);
}
